// update_e_5952824672701
// MI455X (gfx1250) — matrix operand order repaired and checked by reading, not compiled, not run on hardware
//
#include <hip/hip_runtime.h>

// DimeNet++ interaction block ("update_e") for MI455X / gfx1250.
//
// Roofline: ~72 GFLOP of dense GEMM vs ~1.5-2 GB DRAM traffic -> HBM-bound
// (~80us floor at 23.3 TB/s). Dense side uses v_wmma_f32_16x16x32_bf16 with:
//   - weights pre-converted ONCE to bf16 and transposed to [N][K] so a B
//     fragment is exactly 2x global_load_b128 per lane (no cvt, no strided b32)
//   - activations staged as bf16 in LDS so an A fragment is 2x ds_load_b128
// Triplet side streams sbf with non-temporal loads and does the segment-sum
// with hardware global f32 atomics (agg fits in the 192MB L2).

#define H    128   // hidden_channels
#define INTD 64    // int_emb_size
#define BAS  8     // basis_emb_size
#define NSR  42    // num_spherical * num_radial
#define NR   6     // num_radial

typedef __attribute__((ext_vector_type(16))) __bf16 v16bf;
typedef __attribute__((ext_vector_type(8)))  __bf16 v8bf;
typedef __attribute__((ext_vector_type(8)))  float  v8f;
typedef __attribute__((ext_vector_type(2)))  float  f32x2;

__device__ __forceinline__ float swishf(float x) {
  return x / (1.0f + __expf(-x));
}

__device__ __forceinline__ v8f wmma_bf16(v16bf a, v16bf b, v8f c) {
  // D = A(16x32 bf16) * B(32x16 bf16) + C(16x16 f32)
  return __builtin_amdgcn_wmma_f32_16x16x32_bf16(false, a, false, b, (short)0, c,
                                                 false, false);
}

// A fragment (16x32 bf16) from a row-major bf16 tile (LDS). ISA layout:
// VGPR j (j<4): K = half*8 + 2j,2j+1 ; j>=4: K = 16 + half*8 + ...
// => per lane: 8 consecutive bf16 at kb+half*8, 8 more at kb+16+half*8.
__device__ __forceinline__ v16bf a_frag_sbf(const __bf16* buf, int ld, int kb,
                                            int lane) {
  const int m = lane & 15, half = lane >> 4;
  const __bf16* p = buf + m * ld + kb + (half << 3);
  v8bf lo = *(const v8bf*)(p);
  v8bf hi = *(const v8bf*)(p + 16);
  return __builtin_shufflevector(lo, hi, 0, 1, 2, 3, 4, 5, 6, 7, 8, 9, 10, 11,
                                 12, 13, 14, 15);
}

// B fragment (32x16 bf16) from a TRANSPOSED bf16 weight Wt[N][K]. ISA layout:
// VGPR j (j<4): K = half*8 + 2j,2j+1 ; j>=4: K = 16 + half*8 + ... at column n = lane&15
// => per lane: 8 consecutive bf16 at Wt[n0+n][kb + half*8], 8 more at +16.
__device__ __forceinline__ v16bf b_frag_t(const __bf16* __restrict__ bt, int ldk,
                                          int n0, int kb, int lane) {
  const int n = lane & 15, half = lane >> 4;
  const __bf16* p = bt + (size_t)(n0 + n) * ldk + kb + (half << 3);
  v8bf lo = *(const v8bf*)(p);
  v8bf hi = *(const v8bf*)(p + 16);
  return __builtin_shufflevector(lo, hi, 0, 1, 2, 3, 4, 5, 6, 7, 8, 9, 10, 11,
                                 12, 13, 14, 15);
}

// Stage a 16 x W f32 tile (row-major, ld = W) into bf16 LDS, coalesced float4.
template <int W>
__device__ __forceinline__ void stage_tile_bf16(const float* __restrict__ src,
                                                __bf16* dst, int tid) {
  const float4* s4 = (const float4*)src;
#pragma unroll
  for (int i = tid; i < 16 * W / 4; i += 256) {
    float4 v = s4[i];
    __bf16* d = dst + i * 4;
    d[0] = (__bf16)v.x;
    d[1] = (__bf16)v.y;
    d[2] = (__bf16)v.z;
    d[3] = (__bf16)v.w;
  }
}

// ---------------------------------------------------------------------------
// Weight prep: W[K][N] f32 -> Wt[N][K] bf16 (run once per launch; ~0.5MB total)
// ---------------------------------------------------------------------------
__global__ void w2bf_t(const float* __restrict__ W, __bf16* __restrict__ Wt,
                       int K, int N) {
  const int idx = blockIdx.x * blockDim.x + threadIdx.x;
  if (idx >= K * N) return;
  const int k = idx / N, n = idx - k * N;  // coalesced read over n
  Wt[(size_t)n * K + k] = (__bf16)W[idx];
}

// ---------------------------------------------------------------------------
// Kernel 1: per-edge front end.
//   x_ji    = swish(x1 @ Wji + bji)                       [E,H]   -> ws
//   rbf     = (rbf0 @ R1) @ R2
//   x_kjint = swish((swish(x1 @ Wkj + bkj) * rbf) @ Wdown)[E,INT] -> ws
//   rbfe2   = rbf0 @ Wrbf                                 [E,H]   -> ws
// Block = 256 threads (8 waves) = one 16-edge tile; wave w owns cols [16w,16w+16)
// ---------------------------------------------------------------------------
__global__ __launch_bounds__(256) void pre_kernel(
    const float* __restrict__ x1, const float* __restrict__ rbf0,
    const float* __restrict__ R1, const float* __restrict__ R2,
    const __bf16* __restrict__ WjiT, const float* __restrict__ bji,
    const __bf16* __restrict__ WkjT, const float* __restrict__ bkj,
    const __bf16* __restrict__ WdownT, const float* __restrict__ Wrbf,
    float* __restrict__ xji, float* __restrict__ xkjint,
    float* __restrict__ rbfe2) {
  __shared__ __align__(16) __bf16 xbuf[16 * H];  // bf16(x1 tile)
  __shared__ __align__(16) __bf16 hbuf[16 * H];  // swish(x_kj)*rbf
  __shared__ float rbf8[16 * BAS];

  const int r0 = blockIdx.x * 16;
  const int tid = threadIdx.x;
  const int lane = tid & 31, wave = tid >> 5;

  stage_tile_bf16<H>(x1 + (size_t)r0 * H, xbuf, tid);

  // rbf8 = rbf0[tile] @ R1 (16x6 @ 6x8) — tiny, VALU.
  if (tid < 16 * BAS) {
    const int m = tid >> 3, j = tid & 7;
    const float* rr = rbf0 + (size_t)(r0 + m) * NR;
    float acc = 0.f;
#pragma unroll
    for (int k = 0; k < NR; ++k) acc += rr[k] * R1[k * BAS + j];
    rbf8[m * BAS + j] = acc;
  }
  // rbfe2 = rbf0 @ Wrbf (16x6 @ 6x128) — tiny, VALU.
  for (int idx = tid; idx < 16 * H; idx += 256) {
    const int m = idx >> 7, nn = idx & (H - 1);
    const float* rr = rbf0 + (size_t)(r0 + m) * NR;
    float acc = 0.f;
#pragma unroll
    for (int k = 0; k < NR; ++k) acc += rr[k] * Wrbf[k * H + nn];
    rbfe2[(size_t)(r0 + m) * H + nn] = acc;
  }
  __syncthreads();

  const int n0 = wave * 16;
  const int n = lane & 15, half = lane >> 4;

  // Two H x H GEMMs share the A fragment (x1 tile from LDS).
  v8f cji = {}, ckj = {};
#pragma unroll
  for (int kb = 0; kb < H; kb += 32) {
    v16bf a = a_frag_sbf(xbuf, H, kb, lane);
    cji = wmma_bf16(a, b_frag_t(WjiT, H, n0, kb, lane), cji);
    ckj = wmma_bf16(a, b_frag_t(WkjT, H, n0, kb, lane), ckj);
  }
  const float vbji = bji[n0 + n], vbkj = bkj[n0 + n];
#pragma unroll
  for (int i = 0; i < 8; ++i) {
    const int m = (half << 3) + i;
    xji[(size_t)(r0 + m) * H + n0 + n] = swishf(cji[i] + vbji);
    float rb = 0.f;
#pragma unroll
    for (int j = 0; j < BAS; ++j) rb += rbf8[m * BAS + j] * R2[j * H + n0 + n];
    hbuf[m * H + n0 + n] = (__bf16)(swishf(ckj[i] + vbkj) * rb);
  }
  __syncthreads();

  // Down projection h(16x128) @ Wdown(128x64); waves 0..3 (wave-uniform branch)
  if (wave < 4) {
    v8f cd = {};
#pragma unroll
    for (int kb = 0; kb < H; kb += 32)
      cd = wmma_bf16(a_frag_sbf(hbuf, H, kb, lane),
                     b_frag_t(WdownT, H, n0, kb, lane), cd);
#pragma unroll
    for (int i = 0; i < 8; ++i) {
      const int m = (half << 3) + i;
      xkjint[(size_t)(r0 + m) * INTD + n0 + n] = swishf(cd[i]);
    }
  }
}

// ---------------------------------------------------------------------------
// Kernel 2: triplet stream. One thread per triplet.
//   sbf_p = (sbf[t] @ S1) @ S2 ; agg[idx_ji[t]] += x_kjint[idx_kj[t]] * sbf_p
// sbf (336MB) is streamed once -> non-temporal; agg/xkjint live in L2.
// ---------------------------------------------------------------------------
__global__ __launch_bounds__(256) void msg_kernel(
    const float* __restrict__ sbf, const int* __restrict__ idx_kj,
    const int* __restrict__ idx_ji, const float* __restrict__ S1,
    const float* __restrict__ S2, const float* __restrict__ xkjint,
    float* __restrict__ agg, int T) {
  const int t = blockIdx.x * blockDim.x + threadIdx.x;
  if (t >= T) return;
  // sbf rows are 168B => always 8B aligned: use f32x2 non-temporal loads.
  const f32x2* srow2 = (const f32x2*)(sbf + (size_t)t * NSR);
  float s8[BAS];
#pragma unroll
  for (int j = 0; j < BAS; ++j) s8[j] = 0.f;
#pragma unroll
  for (int k2 = 0; k2 < NSR / 2; ++k2) {
    const f32x2 sv = __builtin_nontemporal_load(srow2 + k2);
#pragma unroll
    for (int j = 0; j < BAS; ++j) {
      s8[j] += sv.x * S1[(2 * k2) * BAS + j];
      s8[j] += sv.y * S1[(2 * k2 + 1) * BAS + j];
    }
  }
  const float* xrow = xkjint + (size_t)idx_kj[t] * INTD;
  float* arow = agg + (size_t)idx_ji[t] * INTD;
#pragma unroll 4
  for (int c = 0; c < INTD; ++c) {
    float sp = 0.f;
#pragma unroll
    for (int j = 0; j < BAS; ++j) sp += s8[j] * S2[j * INTD + c];
    atomicAdd(arow + c, xrow[c] * sp);
  }
}

// One residual layer: e1 += swish(swish(bufIn@W1+b1)@W2+b2); bufIn <- bf16(e1).
__device__ __forceinline__ void residual_layer(
    __bf16* bufIn, __bf16* bufTmp, v8f& e1, const __bf16* __restrict__ W1T,
    const float* __restrict__ b1, const __bf16* __restrict__ W2T,
    const float* __restrict__ b2, int n0, int lane) {
  const int n = lane & 15, half = lane >> 4, col = n0 + n;
  v8f c = {};
#pragma unroll
  for (int kb = 0; kb < H; kb += 32)
    c = wmma_bf16(a_frag_sbf(bufIn, H, kb, lane), b_frag_t(W1T, H, n0, kb, lane),
                  c);
  const float vb1 = b1[col];
#pragma unroll
  for (int i = 0; i < 8; ++i)
    bufTmp[((half << 3) + i) * H + col] = (__bf16)swishf(c[i] + vb1);
  __syncthreads();
  v8f c2 = {};
#pragma unroll
  for (int kb = 0; kb < H; kb += 32)
    c2 = wmma_bf16(a_frag_sbf(bufTmp, H, kb, lane),
                   b_frag_t(W2T, H, n0, kb, lane), c2);
  const float vb2 = b2[col];
#pragma unroll
  for (int i = 0; i < 8; ++i) {
    e1[i] += swishf(c2[i] + vb2);
    bufIn[((half << 3) + i) * H + col] = (__bf16)e1[i];
  }
  __syncthreads();
}

// ---------------------------------------------------------------------------
// Kernel 3: per-edge back end (up-proj, cat, before-res, skip, 2x after-res,
// e2). Activations ping-pong through two bf16 LDS tiles; running e1 kept as an
// f32 accumulator fragment per wave.
// ---------------------------------------------------------------------------
__global__ __launch_bounds__(256) void post_kernel(
    const float* __restrict__ x1, const float* __restrict__ xji,
    const float* __restrict__ agg, const __bf16* __restrict__ WupT,
    const __bf16* __restrict__ WcatT, const __bf16* __restrict__ Bw1T,
    const float* __restrict__ Bb1, const __bf16* __restrict__ Bw2T,
    const float* __restrict__ Bb2, const __bf16* __restrict__ WwT,
    const float* __restrict__ bw, const __bf16* __restrict__ A0w1T,
    const float* __restrict__ A0b1, const __bf16* __restrict__ A0w2T,
    const float* __restrict__ A0b2, const __bf16* __restrict__ A1w1T,
    const float* __restrict__ A1b1, const __bf16* __restrict__ A1w2T,
    const float* __restrict__ A1b2, const float* __restrict__ rbfe2,
    float* __restrict__ out_e1, float* __restrict__ out_e2) {
  __shared__ __align__(16) __bf16 bufA[16 * H];
  __shared__ __align__(16) __bf16 bufB[16 * H];

  const int r0 = blockIdx.x * 16;
  const int tid = threadIdx.x;
  const int lane = tid & 31, wave = tid >> 5;
  const int n0 = wave * 16, n = lane & 15, half = lane >> 4;
  const int col = n0 + n;

  // Stage bf16(agg tile) in the first 16x64 of bufB.
  stage_tile_bf16<INTD>(agg + (size_t)r0 * INTD, bufB, tid);
  __syncthreads();

  // up: swish(agg(16x64) @ Wup(64x128)) + x_ji  -> bufA
  v8f c = {};
#pragma unroll
  for (int kb = 0; kb < INTD; kb += 32)
    c = wmma_bf16(a_frag_sbf(bufB, INTD, kb, lane),
                  b_frag_t(WupT, INTD, n0, kb, lane), c);
#pragma unroll
  for (int i = 0; i < 8; ++i) {
    const int m = (half << 3) + i;
    bufA[m * H + col] = (__bf16)(xji[(size_t)(r0 + m) * H + col] + swishf(c[i]));
  }
  __syncthreads();

  // e1 = swish(bufA @ Wcat)  -> bufB
  v8f e1 = {};
  {
    v8f cc = {};
#pragma unroll
    for (int kb = 0; kb < H; kb += 32)
      cc = wmma_bf16(a_frag_sbf(bufA, H, kb, lane),
                     b_frag_t(WcatT, H, n0, kb, lane), cc);
#pragma unroll
    for (int i = 0; i < 8; ++i) {
      e1[i] = swishf(cc[i]);
      bufB[((half << 3) + i) * H + col] = (__bf16)e1[i];
    }
  }
  __syncthreads();

  // num_before_skip = 1 residual layer (in/out bufB)
  residual_layer(bufB, bufA, e1, Bw1T, Bb1, Bw2T, Bb2, n0, lane);

  // e1 = swish(e1 @ Ww + bw) + x1  -> bufA
  {
    v8f cc = {};
#pragma unroll
    for (int kb = 0; kb < H; kb += 32)
      cc = wmma_bf16(a_frag_sbf(bufB, H, kb, lane),
                     b_frag_t(WwT, H, n0, kb, lane), cc);
    const float vb = bw[col];
#pragma unroll
    for (int i = 0; i < 8; ++i) {
      const int m = (half << 3) + i;
      e1[i] = swishf(cc[i] + vb) + x1[(size_t)(r0 + m) * H + col];
      bufA[m * H + col] = (__bf16)e1[i];
    }
  }
  __syncthreads();

  // num_after_skip = 2 residual layers (in/out bufA)
  residual_layer(bufA, bufB, e1, A0w1T, A0b1, A0w2T, A0b2, n0, lane);
  residual_layer(bufA, bufB, e1, A1w1T, A1b1, A1w2T, A1b2, n0, lane);

  // Outputs: e1 and e2 = (rbf0 @ lin_rbf) * e1
#pragma unroll
  for (int i = 0; i < 8; ++i) {
    const int m = (half << 3) + i;
    const size_t o = (size_t)(r0 + m) * H + col;
    out_e1[o] = e1[i];
    out_e2[o] = rbfe2[o] * e1[i];
  }
}

// ---------------------------------------------------------------------------
// Launch. Inputs in setup_inputs() order: x1, rbf0, sbf, idx_kj, idx_ji, q_r,
// then params flattened jax-pytree style (dict keys sorted, lists/tuples in
// order). Workspace: xji[E*H] | xkjint[E*INT] | rbfe2[E*H] | agg[E*INT] f32,
// then the transposed bf16 weights (~360KB).
// ---------------------------------------------------------------------------
extern "C" void kernel_launch(void* const* d_in, const int* in_sizes, int n_in,
                              void* d_out, int out_size, void* d_ws,
                              size_t ws_size, hipStream_t stream) {
  const float* x1   = (const float*)d_in[0];
  const float* rbf0 = (const float*)d_in[1];
  const float* sbf  = (const float*)d_in[2];
  const int* idx_kj = (const int*)d_in[3];
  const int* idx_ji = (const int*)d_in[4];
  // d_in[5] = q_r (unused by forward)
  const float* A0w1 = (const float*)d_in[6];
  const float* A0b1 = (const float*)d_in[7];
  const float* A0w2 = (const float*)d_in[8];
  const float* A0b2 = (const float*)d_in[9];
  const float* A1w1 = (const float*)d_in[10];
  const float* A1b1 = (const float*)d_in[11];
  const float* A1w2 = (const float*)d_in[12];
  const float* A1b2 = (const float*)d_in[13];
  const float* Bw1  = (const float*)d_in[14];
  const float* Bb1  = (const float*)d_in[15];
  const float* Bw2  = (const float*)d_in[16];
  const float* Bb2  = (const float*)d_in[17];
  const float* linb = (const float*)d_in[18];  // lin_b      [128]
  const float* Wcat = (const float*)d_in[19];  // lin_cat    [128,128]
  const float* Wdwn = (const float*)d_in[20];  // lin_down   [128,64]
  const float* bji  = (const float*)d_in[21];  // lin_ji_b   [128]
  const float* Wji  = (const float*)d_in[22];  // lin_ji_w   [128,128]
  const float* bkj  = (const float*)d_in[23];  // lin_kj_b   [128]
  const float* Wkj  = (const float*)d_in[24];  // lin_kj_w   [128,128]
  const float* Wrbf = (const float*)d_in[25];  // lin_rbf    [6,128]
  const float* R1   = (const float*)d_in[26];  // lin_rbf1   [6,8]
  const float* R2   = (const float*)d_in[27];  // lin_rbf2   [8,128]
  const float* S1   = (const float*)d_in[28];  // lin_sbf1   [42,8]
  const float* S2   = (const float*)d_in[29];  // lin_sbf2   [8,64]
  const float* Wup  = (const float*)d_in[30];  // lin_up     [64,128]
  const float* Ww   = (const float*)d_in[31];  // lin_w      [128,128]

  const int E = in_sizes[0] / H;
  const int T = in_sizes[3];

  float* ws     = (float*)d_ws;
  float* xji    = ws;                           // E*H
  float* xkjint = xji + (size_t)E * H;          // E*INTD
  float* rbfe2  = xkjint + (size_t)E * INTD;    // E*H
  float* agg    = rbfe2 + (size_t)E * H;        // E*INTD

  // Transposed bf16 weights after the f32 area (byte offset is 16B aligned).
  __bf16* wt = (__bf16*)(agg + (size_t)E * INTD);
  __bf16* WjiT   = wt;              // [128][128]
  __bf16* WkjT   = WjiT + 16384;    // [128][128]
  __bf16* WdownT = WkjT + 16384;    // [64][128]
  __bf16* WupT   = WdownT + 8192;   // [128][64]
  __bf16* WcatT  = WupT + 8192;     // [128][128]
  __bf16* Bw1T   = WcatT + 16384;
  __bf16* Bw2T   = Bw1T + 16384;
  __bf16* WwT    = Bw2T + 16384;
  __bf16* A0w1T  = WwT + 16384;
  __bf16* A0w2T  = A0w1T + 16384;
  __bf16* A1w1T  = A0w2T + 16384;
  __bf16* A1w2T  = A1w1T + 16384;

  hipMemsetAsync(agg, 0, (size_t)E * INTD * sizeof(float), stream);

  // Weight prep (once per launch, ~0.5MB total).
  w2bf_t<<<64, 256, 0, stream>>>(Wji, WjiT, H, H);
  w2bf_t<<<64, 256, 0, stream>>>(Wkj, WkjT, H, H);
  w2bf_t<<<32, 256, 0, stream>>>(Wdwn, WdownT, H, INTD);
  w2bf_t<<<32, 256, 0, stream>>>(Wup, WupT, INTD, H);
  w2bf_t<<<64, 256, 0, stream>>>(Wcat, WcatT, H, H);
  w2bf_t<<<64, 256, 0, stream>>>(Bw1, Bw1T, H, H);
  w2bf_t<<<64, 256, 0, stream>>>(Bw2, Bw2T, H, H);
  w2bf_t<<<64, 256, 0, stream>>>(Ww, WwT, H, H);
  w2bf_t<<<64, 256, 0, stream>>>(A0w1, A0w1T, H, H);
  w2bf_t<<<64, 256, 0, stream>>>(A0w2, A0w2T, H, H);
  w2bf_t<<<64, 256, 0, stream>>>(A1w1, A1w1T, H, H);
  w2bf_t<<<64, 256, 0, stream>>>(A1w2, A1w2T, H, H);

  pre_kernel<<<E / 16, 256, 0, stream>>>(x1, rbf0, R1, R2, WjiT, bji, WkjT, bkj,
                                         WdownT, Wrbf, xji, xkjint, rbfe2);
  msg_kernel<<<(T + 255) / 256, 256, 0, stream>>>(sbf, idx_kj, idx_ji, S1, S2,
                                                  xkjint, agg, T);

  float* out_e1 = (float*)d_out;
  float* out_e2 = out_e1 + (size_t)E * H;
  post_kernel<<<E / 16, 256, 0, stream>>>(
      x1, xji, agg, WupT, WcatT, Bw1T, Bb1, Bw2T, Bb2, WwT, linb, A0w1T, A0b1,
      A0w2T, A0b2, A1w1T, A1b1, A1w2T, A1b2, rbfe2, out_e1, out_e2);
}
